// MyLSTM_30056181137817
// MI455X (gfx1250) — matrix operand order repaired and checked by reading, not compiled, not run on hardware
//
#include <hip/hip_runtime.h>

typedef __bf16 bf16_t;
typedef __attribute__((ext_vector_type(16))) __bf16 v16bf;
typedef __attribute__((ext_vector_type(8)))  __bf16 v8bf;
typedef __attribute__((ext_vector_type(8)))  float  v8f;

static constexpr int SEQ = 1024;
static constexpr int NB  = 32;        // batch
static constexpr int IN0 = 256;
static constexpr int HID = 512;
static constexpr int K0  = IN0 + HID; // 768
static constexpr int K1  = HID + HID; // 1024

static constexpr size_t W0E   = 4ull * HID * K0;      // bf16 elems
static constexpr size_t W1E   = 4ull * HID * K1;
static constexpr size_t HSLOT = (size_t)NB * HID;     // 16384
static constexpr size_t H0E   = 1025ull * HSLOT;      // h0 history, slots 0..1024
static constexpr int    FLAGS_PER = 1026;

__device__ __forceinline__ unsigned short f2bf(float f) {
  unsigned int u = __float_as_uint(f);
  u += 0x7fffu + ((u >> 16) & 1u);      // round-to-nearest-even
  return (unsigned short)(u >> 16);
}

__device__ __forceinline__ float sigmoidf_(float v) {
  return 1.0f / (1.0f + __expf(-v));
}

// ---------------------------------------------------------------- prep ------
__global__ void lstm_prep(const float* W0f, const float* W0i, const float* W0c, const float* W0o,
                          const float* W1f, const float* W1i, const float* W1c, const float* W1o,
                          unsigned short* Wb0, unsigned short* Wb1,
                          unsigned short* H0, int* flags)
{
  const float* S0[4] = {W0f, W0i, W0c, W0o};
  const float* S1[4] = {W1f, W1i, W1c, W1o};
  const size_t T0 = W0E;
  const size_t T1 = T0 + W1E;
  const size_t T2 = T1 + HSLOT;            // zero h0 slot 0
  const size_t T3 = T2 + 2 * FLAGS_PER;    // flags
  for (size_t i = (size_t)blockIdx.x * blockDim.x + threadIdx.x; i < T3;
       i += (size_t)gridDim.x * blockDim.x) {
    if (i < T0) {
      size_t g = i / ((size_t)HID * K0), r = i % ((size_t)HID * K0);
      Wb0[i] = f2bf(S0[g][r]);
    } else if (i < T1) {
      size_t j = i - T0;
      size_t g = j / ((size_t)HID * K1), r = j % ((size_t)HID * K1);
      Wb1[j] = f2bf(S1[g][r]);
    } else if (i < T2) {
      H0[i - T1] = 0;
    } else {
      size_t j = i - T2;
      flags[j] = (j == 0 || j == (size_t)FLAGS_PER) ? 8 : 0;  // step -1 "done"
    }
  }
}

// --------------------------------------------------------- persistent body --
template <int LAYER, int K, int KS>
__device__ __forceinline__
void lstm_layer_run(const float* __restrict__ x,
                    const unsigned short* __restrict__ Wq,
                    unsigned short* __restrict__ H0,
                    int* flag0, int* flag1,
                    const float* bf_, const float* bi_, const float* bc_, const float* bo_,
                    float* __restrict__ out,
                    unsigned short* lds)
{
  const int wgl  = blockIdx.x & 7;
  const int tid  = threadIdx.x;
  const int wave = tid >> 5;
  const int lane = tid & 31;
  const int lg = lane >> 4, ln = lane & 15;
  const int m0 = (wave >> 2) << 4;               // 0 / 16   (batch tile)
  const int n0 = wgl * 64 + (wave & 3) * 16;     // output-column tile
  const int nn = n0 + ln;

  const float biasF = bf_[nn], biasI = bi_[nn], biasC = bc_[nn], biasO = bo_[nn];

  // B-fragment base per gate: lanes 0-15 -> K=k0..k0+15 at N=nn,
  //                           lanes 16-31 -> K=k0+16..k0+31 at N=nn.
  const bf16_t* Wb = reinterpret_cast<const bf16_t*>(Wq);
  const bf16_t* WgF = Wb + (size_t)(0 * HID + nn) * K + lg * 16;
  const bf16_t* WgI = Wb + (size_t)(1 * HID + nn) * K + lg * 16;
  const bf16_t* WgC = Wb + (size_t)(2 * HID + nn) * K + lg * 16;
  const bf16_t* WgO = Wb + (size_t)(3 * HID + nn) * K + lg * 16;

  v8f cst;                                       // cell state in C/D layout
#pragma unroll
  for (int e = 0; e < 8; ++e) cst[e] = 0.0f;

  constexpr int NKB    = K / 32;
  constexpr int KPC    = K / 8;                  // 8-elem chunks per batch row
  constexpr int NCHUNK = NB * KPC;

  for (int t = 0; t < SEQ; ++t) {
    // ---- wait on producers (one-directional flags, no grid barrier) ----
    if (tid == 0) {
      if (LAYER == 0) {
        while (__hip_atomic_load(&flag0[t], __ATOMIC_ACQUIRE, __HIP_MEMORY_SCOPE_AGENT) < 8)
          __builtin_amdgcn_s_sleep(1);
      } else {
        while (__hip_atomic_load(&flag0[t + 1], __ATOMIC_ACQUIRE, __HIP_MEMORY_SCOPE_AGENT) < 8)
          __builtin_amdgcn_s_sleep(1);
        while (__hip_atomic_load(&flag1[t], __ATOMIC_ACQUIRE, __HIP_MEMORY_SCOPE_AGENT) < 8)
          __builtin_amdgcn_s_sleep(1);
      }
    }
    __syncthreads();

    // ---- stage combined activation [NB][K] (bf16) into LDS ----
    {
      const unsigned short* hsrc = (LAYER == 0) ? (H0 + (size_t)t * HSLOT)
                                                : (H0 + (size_t)(t + 1) * HSLOT);
      for (int cid = tid; cid < NCHUNK; cid += 256) {
        int b  = cid / KPC;
        int kc = (cid % KPC) * 8;
        uint4* dst = (uint4*)&lds[b * K + kc];
        if (LAYER == 0) {
          if (kc < KS) {                         // x_t (fp32 -> bf16)
            const float* s = x + ((size_t)b * SEQ + t) * IN0 + kc;
            float4 f0 = *(const float4*)s;
            float4 f1 = *(const float4*)(s + 4);
            uint4 v;
            v.x = (unsigned)f2bf(f0.x) | ((unsigned)f2bf(f0.y) << 16);
            v.y = (unsigned)f2bf(f0.z) | ((unsigned)f2bf(f0.w) << 16);
            v.z = (unsigned)f2bf(f1.x) | ((unsigned)f2bf(f1.y) << 16);
            v.w = (unsigned)f2bf(f1.z) | ((unsigned)f2bf(f1.w) << 16);
            *dst = v;
          } else {                               // h0(t-1) history, already bf16
            *dst = *(const uint4*)(hsrc + (size_t)b * HID + (kc - KS));
          }
        } else {
          if (kc < KS) {                         // h0(t) fresh from layer0
            *dst = *(const uint4*)(hsrc + (size_t)b * HID + kc);
          } else if (t == 0) {                   // h1(-1) == 0
            *dst = make_uint4(0u, 0u, 0u, 0u);
          } else {                               // h1(t-1) from d_out, fp32->bf16
            const float* s = out + ((size_t)b * SEQ + (t - 1)) * HID + (kc - KS);
            float4 f0 = *(const float4*)s;
            float4 f1 = *(const float4*)(s + 4);
            uint4 v;
            v.x = (unsigned)f2bf(f0.x) | ((unsigned)f2bf(f0.y) << 16);
            v.y = (unsigned)f2bf(f0.z) | ((unsigned)f2bf(f0.w) << 16);
            v.z = (unsigned)f2bf(f1.x) | ((unsigned)f2bf(f1.y) << 16);
            v.w = (unsigned)f2bf(f1.z) | ((unsigned)f2bf(f1.w) << 16);
            *dst = v;
          }
        }
      }
    }
    __syncthreads();

    // ---- 4-gate GEMM for this wave's 16x16 tile ----
    v8f aF, aI, aC, aO;
#pragma unroll
    for (int e = 0; e < 8; ++e) { aF[e] = 0.f; aI[e] = 0.f; aC[e] = 0.f; aO[e] = 0.f; }

    // Launder a zero SGPR offset (fresh opaque value each step): weight loads
    // become loop-variant (no LICM -> no scratch spill) but keep their global
    // address-space provenance, so they lower to global_load_b128 (not flat).
    size_t lz = 0;
    asm volatile("" : "+s"(lz));
    const bf16_t* wF = WgF + lz;
    const bf16_t* wI = WgI + lz;
    const bf16_t* wC = WgC + lz;
    const bf16_t* wO = WgO + lz;

    // A-fragment per lane: chunk0 at k0 + 16*lg, chunk1 at k0 + 16*lg + 8, row m0+ln
    const bf16_t* Arow = reinterpret_cast<const bf16_t*>(lds) + (size_t)(m0 + ln) * K + lg * 16;

#pragma unroll 4
    for (int kb = 0; kb < NKB; ++kb) {
      const int k0 = kb * 32;
      v8bf lo = *(const v8bf*)(Arow + k0);
      v8bf hi = *(const v8bf*)(Arow + k0 + 8);
      v16bf a;
#pragma unroll
      for (int e = 0; e < 8; ++e) { a[e] = lo[e]; a[e + 8] = hi[e]; }
      v16bf bF = *(const v16bf*)(wF + k0);
      v16bf bI = *(const v16bf*)(wI + k0);
      v16bf bC = *(const v16bf*)(wC + k0);
      v16bf bO = *(const v16bf*)(wO + k0);
      __builtin_prefetch(wF + k0 + 128, 0, 3);   // global_prefetch_b8, near caches
      aF = __builtin_amdgcn_wmma_f32_16x16x32_bf16(false, a, false, bF, (short)0, aF, false, false);
      aI = __builtin_amdgcn_wmma_f32_16x16x32_bf16(false, a, false, bI, (short)0, aI, false, false);
      aC = __builtin_amdgcn_wmma_f32_16x16x32_bf16(false, a, false, bC, (short)0, aC, false, false);
      aO = __builtin_amdgcn_wmma_f32_16x16x32_bf16(false, a, false, bO, (short)0, aO, false, false);
    }

    // ---- pointwise LSTM cell; c stays in registers in C/D layout ----
    unsigned short* hdst = H0 + (size_t)(t + 1) * HSLOT;
#pragma unroll
    for (int e = 0; e < 8; ++e) {
      float fg = sigmoidf_(aF[e] + biasF);
      float ig = sigmoidf_(aI[e] + biasI);
      float gg = tanhf(aC[e] + biasC);
      float og = sigmoidf_(aO[e] + biasO);
      float c  = fg * cst[e] + ig * gg;
      cst[e] = c;
      float h = og * tanhf(c);
      int m = m0 + e + 8 * lg;                 // C/D layout: VGPR e, lane group lg
      if (LAYER == 0) {
        hdst[(size_t)m * HID + nn] = f2bf(h);
      } else {
        out[((size_t)m * SEQ + t) * HID + nn] = h;   // final output, fp32
      }
    }

    __threadfence();
    __syncthreads();
    if (tid == 0) {
      int* fl = (LAYER == 0) ? &flag0[t + 1] : &flag1[t + 1];
      __hip_atomic_fetch_add(fl, 1, __ATOMIC_RELEASE, __HIP_MEMORY_SCOPE_AGENT);
    }
  }
}

// ------------------------------------------------------------- kernel -------
__global__ __launch_bounds__(256, 1)
void lstm_persistent(const float* __restrict__ x,
                     const unsigned short* __restrict__ Wb0,
                     const unsigned short* __restrict__ Wb1,
                     unsigned short* __restrict__ H0,
                     int* __restrict__ flags,
                     const float* b0f, const float* b0i, const float* b0c, const float* b0o,
                     const float* b1f, const float* b1i, const float* b1c, const float* b1o,
                     float* __restrict__ out)
{
  __shared__ unsigned short lds[NB * K1];   // 64 KB (layer1); layer0 uses first 48 KB
  int* flag0 = flags;
  int* flag1 = flags + FLAGS_PER;
  if (blockIdx.x < 8) {
    lstm_layer_run<0, K0, IN0>(x, Wb0, H0, flag0, flag1, b0f, b0i, b0c, b0o, out, lds);
  } else {
    lstm_layer_run<1, K1, HID>(x, Wb1, H0, flag0, flag1, b1f, b1i, b1c, b1o, out, lds);
  }
}

// ------------------------------------------------------------- launch -------
extern "C" void kernel_launch(void* const* d_in, const int* in_sizes, int n_in,
                              void* d_out, int out_size, void* d_ws, size_t ws_size,
                              hipStream_t stream)
{
  (void)in_sizes; (void)n_in; (void)out_size; (void)ws_size;
  const float* x   = (const float*)d_in[0];
  const float* W0f = (const float*)d_in[1];
  const float* b0f = (const float*)d_in[2];
  const float* W0i = (const float*)d_in[3];
  const float* b0i = (const float*)d_in[4];
  const float* W0c = (const float*)d_in[5];
  const float* b0c = (const float*)d_in[6];
  const float* W0o = (const float*)d_in[7];
  const float* b0o = (const float*)d_in[8];
  const float* W1f = (const float*)d_in[9];
  const float* b1f = (const float*)d_in[10];
  const float* W1i = (const float*)d_in[11];
  const float* b1i = (const float*)d_in[12];
  const float* W1c = (const float*)d_in[13];
  const float* b1c = (const float*)d_in[14];
  const float* W1o = (const float*)d_in[15];
  const float* b1o = (const float*)d_in[16];

  unsigned short* Wb0 = (unsigned short*)d_ws;        // [4][512][768]  bf16
  unsigned short* Wb1 = Wb0 + W0E;                    // [4][512][1024] bf16
  unsigned short* H0  = Wb1 + W1E;                    // [1025][32][512] bf16
  int* flags          = (int*)(H0 + H0E);             // flag0[1026] | flag1[1026]

  lstm_prep<<<4096, 256, 0, stream>>>(W0f, W0i, W0c, W0o, W1f, W1i, W1c, W1o,
                                      Wb0, Wb1, H0, flags);
  lstm_persistent<<<16, 256, 0, stream>>>(x, Wb0, Wb1, H0, flags,
                                          b0f, b0i, b0c, b0o,
                                          b1f, b1i, b1c, b1o,
                                          (float*)d_out);
}
